// MultiHeadSelfAttention_40767829573917
// MI455X (gfx1250) — matrix operand order repaired and checked by reading, not compiled, not run on hardware
//
#include <hip/hip_runtime.h>
#include <hip/hip_bf16.h>

// ---------------------------------------------------------------------------
// MHA forward for gfx1250 (MI455X), all matmuls via v_wmma_f32_16x16x32_f16.
// Pipeline: fp32->f16 convert -> Q/K/V projection GEMMs -> flash attention
//           (online softmax, LDS transpose staging) -> output projection.
// ---------------------------------------------------------------------------

typedef __attribute__((ext_vector_type(16))) _Float16 v16h;
typedef __attribute__((ext_vector_type(8)))  float    v8f;

union U16h { v16h v; uint4 u[2]; };
union U8h  { uint4 u; _Float16 h[8]; };

__device__ __forceinline__ v8f wmma_f16(v16h a, v16h b, v8f c) {
  // (neg_a, A, neg_b, B, c_mod, C, reuse_a, reuse_b)
  return __builtin_amdgcn_wmma_f32_16x16x32_f16(false, a, false, b, (short)0, c,
                                                false, false);
}

// A fragment (16x32 f16, M x K): lane holds row M = row0+(l&15).
// lanes 0-15: K = {k0..k0+7, k0+16..k0+23}; lanes 16-31: +8.
__device__ __forceinline__ v16h load_a_frag(const _Float16* base, int pitch,
                                            int row0, int k0) {
  const int l   = threadIdx.x & 31;
  const int row = row0 + (l & 15);
  const int kb  = k0 + ((l < 16) ? 0 : 8);
  const _Float16* p = base + (size_t)row * pitch + kb;
  U16h r;
  r.u[0] = *reinterpret_cast<const uint4*>(p);        // K kb..kb+7
  r.u[1] = *reinterpret_cast<const uint4*>(p + 16);   // K kb+16..kb+23
  return r.v;
}

// B fragment (32x16 f16, K x N) where memory stores rows indexed by N with
// contiguous K (i.e. B[k][n] = Mem[n][k]) -- exactly x@W^T / K^T / V^T cases.
// lane holds col N = col0+(l&15); lanes 0-15: K = {k0..k0+7, k0+16..k0+23}; lanes 16-31: +8.
__device__ __forceinline__ v16h load_b_frag(const _Float16* base, int pitch,
                                            int col0, int k0) {
  const int l  = threadIdx.x & 31;
  const int n  = col0 + (l & 15);
  const int kb = k0 + ((l < 16) ? 0 : 8);
  const _Float16* p = base + (size_t)n * pitch + kb;
  U16h r;
  r.u[0] = *reinterpret_cast<const uint4*>(p);        // K kb..kb+7
  r.u[1] = *reinterpret_cast<const uint4*>(p + 16);   // K kb+16..kb+23
  return r.v;
}

// ---------------------------------------------------------------------------
// fp32 -> f16 convert, 4 elements/thread
// ---------------------------------------------------------------------------
__global__ void f32_to_f16_x4(const float* __restrict__ in,
                              _Float16* __restrict__ out, int n4) {
  int i = blockIdx.x * blockDim.x + threadIdx.x;
  if (i < n4) {
    float4 f = reinterpret_cast<const float4*>(in)[i];
    _Float16 h[4] = {(_Float16)f.x, (_Float16)f.y, (_Float16)f.z, (_Float16)f.w};
    reinterpret_cast<uint2*>(out)[i] = *reinterpret_cast<uint2*>(h);
  }
}

// ---------------------------------------------------------------------------
// GEMM: Out[M,N] = A[M,K] @ W[N,K]^T  (M=4096, N=1024, K=1024, f32 accum)
// Block = 256 thr (8 waves, 4Mx2N) -> 128x128 tile; wave -> 32x64 (2x4 frags).
// 8 WMMAs per k-step from 6 fragment loads -> 21.3 FLOP/B from L2; the whole
// f16 working set (A 8MB + W 2MB) lives in the 192MB L2.
// ---------------------------------------------------------------------------
#define GEMM_M 4096
#define GEMM_N 1024
#define GEMM_K 1024

__device__ __forceinline__ void gemm_core(const _Float16* __restrict__ A,
                                          const _Float16* __restrict__ W,
                                          v8f c[2][4], int mBase, int nBase) {
  for (int k0 = 0; k0 < GEMM_K; k0 += 32) {
    v16h a0 = load_a_frag(A, GEMM_K, mBase,      k0);
    v16h a1 = load_a_frag(A, GEMM_K, mBase + 16, k0);
    v16h b0 = load_b_frag(W, GEMM_K, nBase,      k0);
    v16h b1 = load_b_frag(W, GEMM_K, nBase + 16, k0);
    v16h b2 = load_b_frag(W, GEMM_K, nBase + 32, k0);
    v16h b3 = load_b_frag(W, GEMM_K, nBase + 48, k0);
    c[0][0] = wmma_f16(a0, b0, c[0][0]);
    c[0][1] = wmma_f16(a0, b1, c[0][1]);
    c[0][2] = wmma_f16(a0, b2, c[0][2]);
    c[0][3] = wmma_f16(a0, b3, c[0][3]);
    c[1][0] = wmma_f16(a1, b0, c[1][0]);
    c[1][1] = wmma_f16(a1, b1, c[1][1]);
    c[1][2] = wmma_f16(a1, b2, c[1][2]);
    c[1][3] = wmma_f16(a1, b3, c[1][3]);
  }
}

__global__ __launch_bounds__(256) void gemm_xwT_f16out(
    const _Float16* __restrict__ A, const _Float16* __restrict__ W,
    _Float16* __restrict__ O) {
  const int wave  = threadIdx.x >> 5;
  const int lane  = threadIdx.x & 31;
  const int mBase = blockIdx.x * 128 + (wave >> 1) * 32;
  const int nBase = blockIdx.y * 128 + (wave & 1) * 64;
  v8f c[2][4] = {{{}, {}, {}, {}}, {{}, {}, {}, {}}};
  gemm_core(A, W, c, mBase, nBase);
  const int col    = lane & 15;
  const int rowAdd = (lane < 16) ? 0 : 8;
#pragma unroll
  for (int i = 0; i < 2; ++i)
#pragma unroll
    for (int j = 0; j < 4; ++j)
#pragma unroll
      for (int r = 0; r < 8; ++r)
        O[(size_t)(mBase + 16 * i + r + rowAdd) * GEMM_N + nBase + 16 * j + col] =
            (_Float16)c[i][j][r];
}

__global__ __launch_bounds__(256) void gemm_xwT_f32out(
    const _Float16* __restrict__ A, const _Float16* __restrict__ W,
    float* __restrict__ O) {
  const int wave  = threadIdx.x >> 5;
  const int lane  = threadIdx.x & 31;
  const int mBase = blockIdx.x * 128 + (wave >> 1) * 32;
  const int nBase = blockIdx.y * 128 + (wave & 1) * 64;
  v8f c[2][4] = {{{}, {}, {}, {}}, {{}, {}, {}, {}}};
  gemm_core(A, W, c, mBase, nBase);
  const int col    = lane & 15;
  const int rowAdd = (lane < 16) ? 0 : 8;
#pragma unroll
  for (int i = 0; i < 2; ++i)
#pragma unroll
    for (int j = 0; j < 4; ++j)
#pragma unroll
      for (int r = 0; r < 8; ++r)
        O[(size_t)(mBase + 16 * i + r + rowAdd) * GEMM_N + nBase + 16 * j + col] =
            c[i][j][r];
}

// ---------------------------------------------------------------------------
// Flash attention, causal. Q/K/V/O are flat [4096,1024] f16; head h occupies
// cols h*64..h*64+63, batch b occupies rows b*2048..b*2048+2047.
// Grid: (B*H) * (S/128) blocks, 128 threads (4 waves); wave owns 32 q rows
// (two 16-row groups sharing every K/V fragment). Key tiles of 32; V^T and P
// transposed through padded LDS. 16 WMMAs per key tile per wave.
// ---------------------------------------------------------------------------
#define SEQ 2048
#define DMODEL 1024
#define HDIM 64

__global__ __launch_bounds__(128) void attn_kernel(
    const _Float16* __restrict__ Qh, const _Float16* __restrict__ Kh,
    const _Float16* __restrict__ Vh, _Float16* __restrict__ Oh) {
  __shared__ __align__(16) _Float16 Vt[HDIM][40];     // V^T tile: [d][k]
  __shared__ __align__(16) _Float16 Pb[4][32][40];    // per-wave P tile [q][k]

  const int tid  = threadIdx.x;
  const int wave = tid >> 5;
  const int lane = tid & 31;

  const int qTile = blockIdx.x & 15;   // 16 q-tiles (128 rows each) per (b,h)
  const int bh    = blockIdx.x >> 4;   // 0..31
  const int b = bh >> 4, h = bh & 15;

  const size_t base = (size_t)b * SEQ * DMODEL + (size_t)h * HDIM;
  const _Float16* Qb = Qh + base;
  const _Float16* Kb = Kh + base;
  const _Float16* Vb = Vh + base;
  _Float16*       Ob = Oh + base;

  const int qBase = qTile * 128 + wave * 32;

  // Q fragments (invariant over key loop): A-layout, 2 row groups x K-dim 64
  v16h aq[2][2];
#pragma unroll
  for (int i = 0; i < 2; ++i) {
    aq[i][0] = load_a_frag(Qb, DMODEL, qBase + 16 * i, 0);
    aq[i][1] = load_a_frag(Qb, DMODEL, qBase + 16 * i, 32);
  }

  v8f o[2][4] = {{{}, {}, {}, {}}, {{}, {}, {}, {}}};  // [32 q][64 d] accum
  float mrow[2][8], lrow[2][8];
#pragma unroll
  for (int i = 0; i < 2; ++i)
#pragma unroll
    for (int r = 0; r < 8; ++r) { mrow[i][r] = -3.0e38f; lrow[i][r] = 0.0f; }

  const int   col    = lane & 15;
  const int   rowAdd = (lane < 16) ? 0 : 8;
  const float scale  = 0.125f;         // 1/sqrt(64)
  const int   nTiles = (qTile + 1) * 4;

  for (int t = 0; t < nTiles; ++t) {
    const int kc = t * 32;
    __syncthreads();                   // prior-iteration Vt readers done
    {
      // Stage V^T: 128 threads cover 32 rows x 4 col-groups of 16
      const int r  = tid & 31;
      const int cg = tid >> 5;
      const _Float16* vp = Vb + (size_t)(kc + r) * DMODEL + cg * 16;
      U8h w0, w1;
      w0.u = *reinterpret_cast<const uint4*>(vp);
      w1.u = *reinterpret_cast<const uint4*>(vp + 8);
#pragma unroll
      for (int i = 0; i < 8; ++i) {
        Vt[cg * 16 + i][r]     = w0.h[i];
        Vt[cg * 16 + 8 + i][r] = w1.h[i];
      }
    }
    __syncthreads();

    // K-tile B fragments: 2 key groups x 2 d-steps (shared by both row groups)
    v16h bk[2][2];
#pragma unroll
    for (int jk = 0; jk < 2; ++jk) {
      bk[jk][0] = load_b_frag(Kb, DMODEL, kc + 16 * jk, 0);
      bk[jk][1] = load_b_frag(Kb, DMODEL, kc + 16 * jk, 32);
    }

    // S = Q @ K^T : 8 WMMAs (2 row groups x 2 key groups x 2 k-steps)
    v8f s[2][2] = {{{}, {}}, {{}, {}}};
#pragma unroll
    for (int i = 0; i < 2; ++i)
#pragma unroll
      for (int jk = 0; jk < 2; ++jk) {
        s[i][jk] = wmma_f16(aq[i][0], bk[jk][0], s[i][jk]);
        s[i][jk] = wmma_f16(aq[i][1], bk[jk][1], s[i][jk]);
      }

    // Online softmax per row group: scale, causal mask, 16-lane reductions
#pragma unroll
    for (int i = 0; i < 2; ++i) {
#pragma unroll
      for (int r = 0; r < 8; ++r) {
        const int qrow = qBase + 16 * i + r + rowAdd;
        float v0 = s[i][0][r] * scale;
        float v1 = s[i][1][r] * scale;
        if (kc + col > qrow)      v0 = -1.0e30f;
        if (kc + 16 + col > qrow) v1 = -1.0e30f;
        float mx = fmaxf(v0, v1);
#pragma unroll
        for (int off = 1; off < 16; off <<= 1)
          mx = fmaxf(mx, __shfl_xor(mx, off, 32));
        const float nm = fmaxf(mrow[i][r], mx);
        const float sf = __expf(mrow[i][r] - nm);
        mrow[i][r] = nm;
        const float p0 = __expf(v0 - nm);
        const float p1 = __expf(v1 - nm);
        float rs = p0 + p1;
#pragma unroll
        for (int off = 1; off < 16; off <<= 1)
          rs += __shfl_xor(rs, off, 32);
        lrow[i][r] = lrow[i][r] * sf + rs;
        o[i][0][r] *= sf; o[i][1][r] *= sf;
        o[i][2][r] *= sf; o[i][3][r] *= sf;
        Pb[wave][16 * i + r + rowAdd][col]      = (_Float16)p0;  // C->LDS
        Pb[wave][16 * i + r + rowAdd][col + 16] = (_Float16)p1;
      }
    }
    // No barrier needed: P tile is wave-private, and DS ops within one wave
    // complete in order (DScnt) -- backend inserts s_wait_dscnt for the RAW.

    // O += P @ V : A from LDS P (A-layout), B from LDS V^T (reused across i)
    v16h bv[4];
#pragma unroll
    for (int j = 0; j < 4; ++j)
      bv[j] = load_b_frag(&Vt[0][0], 40, 16 * j, 0);
#pragma unroll
    for (int i = 0; i < 2; ++i) {
      const v16h ap = load_a_frag(&Pb[wave][0][0], 40, 16 * i, 0);
      o[i][0] = wmma_f16(ap, bv[0], o[i][0]);
      o[i][1] = wmma_f16(ap, bv[1], o[i][1]);
      o[i][2] = wmma_f16(ap, bv[2], o[i][2]);
      o[i][3] = wmma_f16(ap, bv[3], o[i][3]);
    }
  }

  // Normalize and write attention output (f16)
#pragma unroll
  for (int i = 0; i < 2; ++i)
#pragma unroll
    for (int r = 0; r < 8; ++r) {
      const float inv = 1.0f / lrow[i][r];
      const size_t row = (size_t)(qBase + 16 * i + r + rowAdd) * DMODEL;
#pragma unroll
      for (int j = 0; j < 4; ++j)
        Ob[row + j * 16 + col] = (_Float16)(o[i][j][r] * inv);
    }
}

// ---------------------------------------------------------------------------
// Host launcher
// ---------------------------------------------------------------------------
extern "C" void kernel_launch(void* const* d_in, const int* in_sizes, int n_in,
                              void* d_out, int out_size, void* d_ws,
                              size_t ws_size, hipStream_t stream) {
  (void)in_sizes; (void)n_in; (void)out_size; (void)ws_size;
  const float* x  = (const float*)d_in[0];
  const float* Wq = (const float*)d_in[1];
  const float* Wk = (const float*)d_in[2];
  const float* Wv = (const float*)d_in[3];
  const float* Wo = (const float*)d_in[4];
  float* out = (float*)d_out;

  char* ws = (char*)d_ws;
  const size_t MiB = 1ull << 20;
  _Float16* xh  = (_Float16*)(ws + 0 * MiB);   // 8 MiB  [4096,1024]
  _Float16* Wqh = (_Float16*)(ws + 8 * MiB);   // 2 MiB each
  _Float16* Wkh = (_Float16*)(ws + 10 * MiB);
  _Float16* Wvh = (_Float16*)(ws + 12 * MiB);
  _Float16* Woh = (_Float16*)(ws + 14 * MiB);
  _Float16* Qh  = (_Float16*)(ws + 16 * MiB);  // 8 MiB each
  _Float16* Kh  = (_Float16*)(ws + 24 * MiB);
  _Float16* Vh  = (_Float16*)(ws + 32 * MiB);
  _Float16* Ah  = (_Float16*)(ws + 40 * MiB);  // attention output, pre-Wo

  auto cvt = [&](const float* src, _Float16* dst, int n) {
    int n4 = n / 4;
    f32_to_f16_x4<<<(n4 + 255) / 256, 256, 0, stream>>>(src, dst, n4);
  };
  cvt(x,  xh,  GEMM_M * GEMM_K);
  cvt(Wq, Wqh, GEMM_N * GEMM_K);
  cvt(Wk, Wkh, GEMM_N * GEMM_K);
  cvt(Wv, Wvh, GEMM_N * GEMM_K);
  cvt(Wo, Woh, GEMM_N * GEMM_K);

  dim3 gg(GEMM_M / 128, GEMM_N / 128);
  gemm_xwT_f16out<<<gg, 256, 0, stream>>>(xh, Wqh, Qh);
  gemm_xwT_f16out<<<gg, 256, 0, stream>>>(xh, Wkh, Kh);
  gemm_xwT_f16out<<<gg, 256, 0, stream>>>(xh, Wvh, Vh);

  attn_kernel<<<dim3(32 * 16), 128, 0, stream>>>(Qh, Kh, Vh, Ah);

  gemm_xwT_f32out<<<gg, 256, 0, stream>>>(Ah, Woh, out);
}
